// GRUModel_1219770712109
// MI455X (gfx1250) — matrix operand order repaired and checked by reading, not compiled, not run on hardware
//
#include <hip/hip_runtime.h>
#include <math.h>

#define B_   256
#define T_   1024
#define F_   37
#define U_   64
#define NW   12        // 12 waves: one 16-col WMMA tile each over 3U=192 columns
#define BT   (NW * 32) // 384 threads
#define ROWS 16        // batch rows per workgroup -> 16 workgroups
#define VS   72        // vt/hbf row stride in ushorts (bank-conflict padding, 16B aligned)
#define PS   196       // xpbuf/hpbuf row stride in floats (8*196 % 64 == 32)

typedef __attribute__((ext_vector_type(16))) __bf16 v16bf;
typedef __attribute__((ext_vector_type(8)))  float  v8f;

union Frag {
  uint4          q[2];
  v16bf          v;
  unsigned short u[16];
};

// native hardware f32 -> bf16 conversion (RNE)
__device__ __forceinline__ unsigned short bf16u(float f) {
  return __builtin_bit_cast(unsigned short, (__bf16)f);
}
__device__ __forceinline__ unsigned int pk_bf16(float a, float b) {
  return (unsigned int)bf16u(a) | ((unsigned int)bf16u(b) << 16);
}
// fast transcendental path: v_exp_f32 + v_rcp_f32 (no IEEE divide expansion)
__device__ __forceinline__ float sigmoidf_(float x) {
  return __builtin_amdgcn_rcpf(1.0f + __expf(-x));
}
__device__ __forceinline__ float tanhf_(float x) {
  return 2.0f * sigmoidf_(2.0f * x) - 1.0f;
}
__device__ __forceinline__ float wave_sum32(float v) {
  #pragma unroll
  for (int off = 16; off > 0; off >>= 1) v += __shfl_xor(v, off, 32);
  return v;
}
// LDS-only split barrier: do NOT fence global (keeps rv prefetch loads and the
// out store in flight across the barrier, unlike __syncthreads' full fence).
__device__ __forceinline__ void lds_barrier() {
  asm volatile("s_wait_dscnt 0x0\n\t"
               "s_barrier_signal -1\n\t"
               "s_barrier_wait -1" ::: "memory");
}

__global__ __launch_bounds__(BT)
__attribute__((amdgpu_waves_per_eu(1)))   // occupancy is grid-limited (16 WGs); let RA use VGPRs freely
void gru_fused_wmma(
    const float* __restrict__ values,   // [B,T,F]
    const int*   __restrict__ lengths,  // [B]
    const float* __restrict__ Wk,       // [F,3U]
    const float* __restrict__ Wr,       // [U,3U]
    const float* __restrict__ bias,     // [2,3U]
    const float* __restrict__ dense_w,  // [U,1]
    const float* __restrict__ dense_b,  // [1]
    float*       __restrict__ out)      // [B,T,1]
{
  __shared__ unsigned short vt[ROWS * VS]  __attribute__((aligned(16))); // bf16 values tile (K pad->64)
  __shared__ unsigned short hbf[ROWS * VS] __attribute__((aligned(16))); // bf16 hidden state (WMMA A source)
  __shared__ float xpbuf[ROWS * PS];    // x-projection tile (192 cols)
  __shared__ float hpbuf[ROWS * PS];    // recurrent tile   (192 cols)

  const int tid  = threadIdx.x;
  const int lane = tid & 31;
  const int lm   = lane & 15;
  const int lh   = lane >> 4;
  const int w    = tid >> 5;            // wave id 0..11 -> N-tile
  const int b0   = blockIdx.x * ROWS;
  const int colN = 16 * w + lm;         // global column in [0,192)

  // ---------- one-time setup ----------
  for (int e = tid; e < ROWS * VS; e += BT) { vt[e] = 0; hbf[e] = 0; }

  // Weight B-fragments, resident in VGPRs for all 1024 steps.
  // B layout (16-bit, 32x16): element i = B[32*s + (i<8 ? 8*lh + i : 16 + 8*lh + i-8)][colN], N = lm.
  Frag kf[2], rf[2];
  #pragma unroll
  for (int s = 0; s < 2; ++s) {
    #pragma unroll
    for (int i = 0; i < 16; ++i) {
      int K = 32 * s + ((i < 8) ? (8 * lh + i) : (16 + 8 * lh + (i - 8)));
      kf[s].u[i] = (K < F_) ? bf16u(Wk[K * 192 + colN]) : (unsigned short)0;
      rf[s].u[i] = bf16u(Wr[K * 192 + colN]);
    }
  }
  const float xb = bias[colN];          // bias[0][col]
  const float hb = bias[192 + colN];    // bias[1][col]

  // ---------- gate-phase invariants: thread <-> element-pair mapping ----------
  // 512 pairs over 384 threads: chunk A pair=tid (wave w owns row w),
  // chunk B pair=tid+384 for tid<128 (waves 0..3 own rows 12..15).
  const int  uu   = lane * 2;
  const int  rowA = w;
  const int  rowB = 12 + w;
  const bool hasB = (tid < 128);
  const int  pbA  = rowA * PS + uu;
  const int  pbB  = rowB * PS + uu;
  const int  hbA  = rowA * VS + uu;
  const int  hbB  = rowB * VS + uu;
  const int  lenA = lengths[b0 + rowA];
  const int  lenB = lengths[b0 + rowB];
  const float dw0 = dense_w[uu];
  const float dw1 = dense_w[uu + 1];
  const float dbv = dense_b[0];
  float* const outA = out + (size_t)(b0 + rowA) * T_;
  float* const outB = out + (size_t)(b0 + rowB) * T_;
  float hA0 = 0.f, hA1 = 0.f, hB0 = 0.f, hB1 = 0.f;   // hidden state lives in registers

  // ---------- values register pipeline (16*37 = 592 elems, <=2 per thread) ----------
  const int  e0   = tid;                        // always < 592
  const int  e1   = tid + BT;
  const bool has1 = (e1 < ROWS * F_);           // tid < 208
  const int  r0 = e0 / F_, k0 = e0 - r0 * F_;
  const int  r1 = has1 ? (e1 / F_) : 0;
  const int  k1 = has1 ? (e1 - r1 * F_) : 0;
  const float* p0 = values + (size_t)(b0 + r0) * T_ * F_ + k0;
  const float* p1 = values + (size_t)(b0 + r1) * T_ * F_ + k1;
  const int  d0 = r0 * VS + k0;
  const int  d1 = r1 * VS + k1;

  float rv0 = p0[0];                            // t = 0
  float rv1 = has1 ? p1[0] : 0.f;
  vt[d0] = bf16u(rv0);
  if (has1) vt[d1] = bf16u(rv1);
  rv0 = p0[F_];                                 // preload t = 1
  if (has1) rv1 = p1[F_];
  __syncthreads();                              // once, full fence is fine here

  const int ib = lm * VS + lh * 8;              // A-fragment base (loop-invariant)

  for (int t = 0; t < T_; ++t) {
    // ---- issue all 8 fragment loads; sched_barrier keeps them before the WMMAs
    //      so all 4 fragments are simultaneously live (distinct VGPRs, one wait) ----
    Frag fh0, fh1, fa0, fa1;
    fh0.q[0] = *(const uint4*)(hbf + ib);
    fh0.q[1] = *(const uint4*)(hbf + ib + 16);
    fh1.q[0] = *(const uint4*)(hbf + ib + 32);
    fh1.q[1] = *(const uint4*)(hbf + ib + 48);
    fa0.q[0] = *(const uint4*)(vt  + ib);
    fa0.q[1] = *(const uint4*)(vt  + ib + 16);
    fa1.q[0] = *(const uint4*)(vt  + ib + 32);
    fa1.q[1] = *(const uint4*)(vt  + ib + 48);
    __builtin_amdgcn_sched_barrier(0);

    v8f hacc = {hb, hb, hb, hb, hb, hb, hb, hb};
    v8f xacc = {xb, xb, xb, xb, xb, xb, xb, xb};
    // interleave the two accumulate chains: h-chain (true recurrence) first,
    // x-chain ops fill the D->C dependency shadow between h-chain WMMAs
    hacc = __builtin_amdgcn_wmma_f32_16x16x32_bf16(false, fh0.v, false, rf[0].v,
                                                   (short)0, hacc, false, false);
    xacc = __builtin_amdgcn_wmma_f32_16x16x32_bf16(false, fa0.v, false, kf[0].v,
                                                   (short)0, xacc, false, false);
    hacc = __builtin_amdgcn_wmma_f32_16x16x32_bf16(false, fh1.v, false, rf[1].v,
                                                   (short)0, hacc, false, false);
    xacc = __builtin_amdgcn_wmma_f32_16x16x32_bf16(false, fa1.v, false, kf[1].v,
                                                   (short)0, xacc, false, false);

    // ---- branchless scatter of both tiles (C layout: M = r2 + 8*lh, N = colN) ----
    #pragma unroll
    for (int r2 = 0; r2 < 8; ++r2) {
      int M = r2 + 8 * lh;
      hpbuf[M * PS + colN] = hacc[r2];
      xpbuf[M * PS + colN] = xacc[r2];
    }
    lds_barrier();                      // D: tiles visible

    // ---- gate update on register-resident h, fused dense head ----
    auto gate_pair = [&](int pb, int hbi, int len, float& h0, float& h1) -> float {
      float z0 = sigmoidf_(xpbuf[pb]       + hpbuf[pb]);
      float z1 = sigmoidf_(xpbuf[pb + 1]   + hpbuf[pb + 1]);
      float g0 = sigmoidf_(xpbuf[pb + 64]  + hpbuf[pb + 64]);
      float g1 = sigmoidf_(xpbuf[pb + 65]  + hpbuf[pb + 65]);
      float c0 = tanhf_(xpbuf[pb + 128] + g0 * hpbuf[pb + 128]);
      float c1 = tanhf_(xpbuf[pb + 129] + g1 * hpbuf[pb + 129]);
      float hn0 = z0 * h0 + (1.f - z0) * c0;
      float hn1 = z1 * h1 + (1.f - z1) * c1;
      bool keep = (t < len);            // masked steps keep previous state
      h0 = keep ? hn0 : h0;
      h1 = keep ? hn1 : h1;
      *(unsigned int*)(hbf + hbi) = pk_bf16(h0, h1);  // packed bf16 store
      return h0 * dw0 + h1 * dw1;       // dense-head contribution
    };

    float accA = wave_sum32(gate_pair(pbA, hbA, lenA, hA0, hA1));
    if (lane == 0) outA[t] = sigmoidf_(dbv + accA);
    if (hasB) {
      float accB = wave_sum32(gate_pair(pbB, hbB, lenB, hB0, hB1));
      if (lane == 0) outB[t] = sigmoidf_(dbv + accB);
    }

    // ---- stage vt(t+1) from registers, then issue loads for t+2 ----
    vt[d0] = bf16u(rv0);
    if (has1) vt[d1] = bf16u(rv1);
    int tl = (t + 2 < T_) ? (t + 2) : (T_ - 1);
    rv0 = p0[(size_t)tl * F_];
    if (has1) rv1 = p1[(size_t)tl * F_];
    lds_barrier();                      // F: h(bf16) and vt(t+1) visible
  }
}

extern "C" void kernel_launch(void* const* d_in, const int* in_sizes, int n_in,
                              void* d_out, int out_size, void* d_ws, size_t ws_size,
                              hipStream_t stream) {
  // setup_inputs order: 0 demo, 1 times, 2 values, 3 measurements, 4 lengths,
  //                     5 kernel, 6 recurrent_kernel, 7 bias, 8 dense_w, 9 dense_b
  const float* values  = (const float*)d_in[2];
  const int*   lengths = (const int*)  d_in[4];
  const float* Wk      = (const float*)d_in[5];
  const float* Wr      = (const float*)d_in[6];
  const float* bias    = (const float*)d_in[7];
  const float* dwp     = (const float*)d_in[8];
  const float* dbp     = (const float*)d_in[9];
  float* out = (float*)d_out;

  gru_fused_wmma<<<dim3(B_ / ROWS), dim3(BT), 0, stream>>>(
      values, lengths, Wk, Wr, bias, dwp, dbp, out);

  (void)in_sizes; (void)n_in; (void)out_size; (void)d_ws; (void)ws_size;
}
